// LeafLayer_9517647527944
// MI455X (gfx1250) — matrix operand order repaired and checked by reading, not compiled, not run on hardware
//
#include <hip/hip_runtime.h>
#include <hip/hip_bf16.h>

// MI455X / gfx1250: wave32, WMMA 16x16x32 f16 -> f32.
typedef __attribute__((ext_vector_type(16))) _Float16 v16h;
typedef __attribute__((ext_vector_type(8)))  _Float16 v8h;
typedef __attribute__((ext_vector_type(8)))  float    v8f;

#define BB 8
#define NN 2048
#define MM 128
#define HH 128

union AF { v16h v; v8h h[2]; };

__device__ __forceinline__ v8f wmma_f16(v16h a, v16h b, v8f c) {
  // 8 args: (neg_a, A, neg_b, B, c_mod, C, reuse_a, reuse_b)
  return __builtin_amdgcn_wmma_f32_16x16x32_f16(false, a, false, b, (short)0, c,
                                                false, false);
}

// ---------------------------------------------------------------- convert
__global__ void cvt_f32_f16(const float* __restrict__ src,
                            _Float16* __restrict__ dst, int n) {
  for (int i = blockIdx.x * blockDim.x + threadIdx.x; i < n;
       i += gridDim.x * blockDim.x)
    dst[i] = (_Float16)src[i];
}

// ------------------------------------------------ C = relu(A[R,128] @ W^T + b)
// W is [128 out, 128 in] row-major, so B-matrix column n == W row n (contig K).
// MODE 0: f16 [R,128] row-major (LDS-staged, packed b128 stores)
// MODE 1: f16 transposed per batch out[b][n][i] (one packed v8h store per lane)
// MODE 2: f32 [R,128] row-major (LDS-staged, packed b128 stores)
template <int MODE>
__global__ __launch_bounds__(256)
void gemm128(const _Float16* __restrict__ A16, const _Float16* __restrict__ W16,
             const float* __restrict__ bias, _Float16* __restrict__ out16,
             float* __restrict__ out32) {
  __shared__ __align__(32) unsigned char smem[8192];
  _Float16* ldsA = (_Float16*)smem;                 // 16x128 f16 A tile (4KB)

  const int r0   = blockIdx.x * 16;
  const int lane = threadIdx.x & 31;
  const int wave = threadIdx.x >> 5;
  const int l15  = lane & 15;
  const int hf   = lane >> 4;

  // Stage the 16x128 f16 A tile in LDS (256 threads x 16B).
  ((v8h*)ldsA)[threadIdx.x] = ((const v8h*)(A16 + (size_t)r0 * 128))[threadIdx.x];
  __syncthreads();

  const int n0 = wave * 16;           // 8 waves cover 128 output columns
  v8f acc = {};
#pragma unroll
  for (int c = 0; c < 4; ++c) {       // K = 128 in 4 chunks of 32
    AF a;                             // A frag: lane<16 K{0-7,16-23}, lane>=16 K{8-15,24-31}
    a.h[0] = *(const v8h*)(ldsA + l15 * 128 + c * 32 + hf * 8);
    a.h[1] = *(const v8h*)(ldsA + l15 * 128 + c * 32 + 16 + hf * 8);
    // B 32x16 frag: column n = W row (n0+l15), same K order as A frag
    AF bfr_u;
    bfr_u.h[0] = *(const v8h*)(W16 + (size_t)(n0 + l15) * 128 + c * 32 + hf * 8);
    bfr_u.h[1] = *(const v8h*)(W16 + (size_t)(n0 + l15) * 128 + c * 32 + 16 + hf * 8);
    v16h bfr = bfr_u.v;
    acc = wmma_f16(a.v, bfr, acc);
  }

  const float bcol = bias[n0 + l15];
  const int   n    = n0 + l15;

  if (MODE == 1) {
    // g = 0..7 are contiguous along i -> single packed 16B store per lane.
    v8h pack;
#pragma unroll
    for (int g = 0; g < 8; ++g)
      pack[g] = (_Float16)fmaxf(acc[g] + bcol, 0.0f);
    const int b = r0 >> 11;                       // N = 2048
    const int i = (r0 & (NN - 1)) + 8 * hf;
    *(v8h*)(out16 + (((size_t)b * HH + n) << 11) + i) = pack;
    return;
  }

  // Stage output tile in LDS (C layout -> row major), then packed b128 stores.
  __syncthreads();                                 // protect ldsA before reuse
  if (MODE == 0) {
    _Float16* ot = (_Float16*)smem;                // 16x128 f16 (4KB)
#pragma unroll
    for (int g = 0; g < 8; ++g)
      ot[(g + 8 * hf) * 128 + n] = (_Float16)fmaxf(acc[g] + bcol, 0.0f);
    __syncthreads();
    // 16x128 tile is contiguous in out16: one b128 store per thread.
    ((v8h*)(out16 + (size_t)r0 * 128))[threadIdx.x] = ((const v8h*)ot)[threadIdx.x];
  } else {                                         // MODE == 2, f32 out
    float* ot = (float*)smem;                      // 16x128 f32 (8KB)
#pragma unroll
    for (int g = 0; g < 8; ++g)
      ot[(g + 8 * hf) * 128 + n] = fmaxf(acc[g] + bcol, 0.0f);
    __syncthreads();
    ((v8f*)(out32 + (size_t)r0 * 128))[threadIdx.x] = ((const v8f*)ot)[threadIdx.x];
  }
}

// ------------------------------------------------ flash attention
// One wave owns 16 query rows; loops over 64-wide key/value blocks.
// S = (q @ k^T) * d ; online softmax ; O = sum(P @ V).
__global__ __launch_bounds__(128)
void attn_flash(const _Float16* __restrict__ q16, const _Float16* __restrict__ k16,
                const _Float16* __restrict__ vT16, const float* __restrict__ dmask,
                _Float16* __restrict__ o16) {
  __shared__ __align__(32) _Float16 ldsP[4][16 * 64];   // per-wave P buffer (8KB)
  __shared__ __align__(32) _Float16 ldsO[4][16 * 128];  // per-wave O staging (16KB)
  const int lane = threadIdx.x & 31;
  const int wave = threadIdx.x >> 5;
  const int l15  = lane & 15;
  const int hf   = lane >> 4;
  const int b    = blockIdx.y;
  const int i0   = (blockIdx.x * 4 + wave) * 16;

  // Preload q A-fragments for K = 0..127 (kept in VGPRs all kernel).
  AF qf[4];
#pragma unroll
  for (int c = 0; c < 4; ++c) {
    const _Float16* p = q16 + (size_t)(b * NN + i0 + l15) * HH + c * 32 + hf * 8;
    qf[c].h[0] = *(const v8h*)p;
    qf[c].h[1] = *(const v8h*)(p + 16);
  }

  v8f zero = {};
  v8f O[8];
  float m[8], s[8], alpha[8];
#pragma unroll
  for (int t = 0; t < 8; ++t) O[t] = zero;
#pragma unroll
  for (int g = 0; g < 8; ++g) { m[g] = -3.0e38f; s[g] = 0.0f; }

  _Float16* P = ldsP[wave];

  for (int j0 = 0; j0 < NN; j0 += 64) {
    // ---- S[16x64] = q @ k^T (four 16-wide column tiles, K = 128)
    v8f S[4];
#pragma unroll
    for (int t = 0; t < 4; ++t) S[t] = zero;
#pragma unroll
    for (int c = 0; c < 4; ++c) {
      const _Float16* kp = k16 + (size_t)(b * NN + j0 + l15) * HH + c * 32 + hf * 8;
#pragma unroll
      for (int t = 0; t < 4; ++t) {
        AF kb;
        kb.h[0] = *(const v8h*)(kp + (size_t)(t * 16) * HH);
        kb.h[1] = *(const v8h*)(kp + (size_t)(t * 16) * HH + 16);
        S[t] = wmma_f16(qf[c].v, kb.v, S[t]);
      }
    }

    // ---- multiply by d (single streaming read of the 128MB tensor)
    const float* dp = dmask + (size_t)(b * NN + i0 + 8 * hf) * NN + j0 + l15;
    if (j0 + 64 < NN) __builtin_prefetch(dp + 64, 0, 1);   // global_prefetch_b8

    // ---- online softmax, emit P = exp(x - m_new) into LDS
    bool grew = false;
#pragma unroll
    for (int g = 0; g < 8; ++g) {
      const float x0 = S[0][g] * dp[(size_t)g * NN];
      const float x1 = S[1][g] * dp[(size_t)g * NN + 16];
      const float x2 = S[2][g] * dp[(size_t)g * NN + 32];
      const float x3 = S[3][g] * dp[(size_t)g * NN + 48];
      float mx = fmaxf(fmaxf(x0, x1), fmaxf(x2, x3));
#pragma unroll
      for (int msk = 1; msk < 16; msk <<= 1)
        mx = fmaxf(mx, __shfl_xor(mx, msk, 32));   // stays in 16-lane half
      const float mnew = fmaxf(m[g], mx);
      grew = grew || (mnew > m[g]);
      alpha[g] = __expf(m[g] - mnew);
      const float e0 = __expf(x0 - mnew);
      const float e1 = __expf(x1 - mnew);
      const float e2 = __expf(x2 - mnew);
      const float e3 = __expf(x3 - mnew);
      float ls = (e0 + e1) + (e2 + e3);
#pragma unroll
      for (int msk = 1; msk < 16; msk <<= 1) ls += __shfl_xor(ls, msk, 32);
      s[g] = s[g] * alpha[g] + ls;
      m[g] = mnew;
      const int prow = (g + 8 * hf) * 64 + l15;    // C-layout -> row major
      P[prow]      = (_Float16)e0;
      P[prow + 16] = (_Float16)e1;
      P[prow + 32] = (_Float16)e2;
      P[prow + 48] = (_Float16)e3;
    }

    // ---- reload P as two A 16x32 fragments (same wave: LDS kept in order)
    AF pa[2];
#pragma unroll
    for (int c = 0; c < 2; ++c) {
      pa[c].v = *(const v16h*)(P + l15 * 64 + c * 32 + hf * 16);
    }

    // ---- rescale O only when some row's max actually grew (alpha != 1)
    if (__any(grew)) {
#pragma unroll
      for (int ct = 0; ct < 8; ++ct)
#pragma unroll
        for (int g = 0; g < 8; ++g) O[ct][g] *= alpha[g];
    }

    // ---- O += P @ V   (V from transposed store: contiguous K)
#pragma unroll
    for (int ct = 0; ct < 8; ++ct) {
      const _Float16* vp =
          vT16 + (size_t)(b * HH + ct * 16 + l15) * NN + j0 + hf * 16;
      O[ct] = wmma_f16(pa[0].v, *(const v16h*)vp, O[ct]);
      O[ct] = wmma_f16(pa[1].v, *(const v16h*)(vp + 32), O[ct]);
    }
  }

  // ---- normalize, stage 16x128 tile in per-wave LDS, packed b128 stores
#pragma unroll
  for (int g = 0; g < 8; ++g) s[g] = 1.0f / s[g];
  _Float16* ot = ldsO[wave];
#pragma unroll
  for (int ct = 0; ct < 8; ++ct) {
#pragma unroll
    for (int g = 0; g < 8; ++g)
      ot[(g + 8 * hf) * 128 + ct * 16 + l15] = (_Float16)(O[ct][g] * s[g]);
  }
  // Tile rows i0..i0+15 are contiguous in o16: 8 x 16B per lane.
  const size_t base = (size_t)(b * NN + i0) * HH;
#pragma unroll
  for (int u = 0; u < 8; ++u) {
    const int idx = (u * 32 + lane) * 8;
    *(v8h*)(o16 + base + idx) = *(const v8h*)(ot + idx);
  }
}

// ---------------------------------------------------------------- launcher
extern "C" void kernel_launch(void* const* d_in, const int* in_sizes, int n_in,
                              void* d_out, int out_size, void* d_ws, size_t ws_size,
                              hipStream_t stream) {
  const float* d  = (const float*)d_in[0];
  const float* h  = (const float*)d_in[1];
  const float* Wv = (const float*)d_in[2];
  const float* bv = (const float*)d_in[3];
  const float* Wk = (const float*)d_in[4];
  const float* bk = (const float*)d_in[5];
  const float* Wq = (const float*)d_in[6];
  const float* bq = (const float*)d_in[7];
  const float* Wo = (const float*)d_in[8];
  const float* bo = (const float*)d_in[9];
  float* out = (float*)d_out;

  // Workspace layout (f16): ~20.2 MB total
  _Float16* h16  = (_Float16*)d_ws;                       // [B*N, M]
  _Float16* Wq16 = h16  + (size_t)BB * NN * MM;           // [H, M]
  _Float16* Wk16 = Wq16 + (size_t)HH * MM;
  _Float16* Wv16 = Wk16 + (size_t)HH * MM;
  _Float16* Wo16 = Wv16 + (size_t)HH * MM;                // [M, H]
  _Float16* q16  = Wo16 + (size_t)MM * HH;                // [B*N, H]
  _Float16* k16  = q16  + (size_t)BB * NN * HH;           // [B*N, H]
  _Float16* vT16 = k16  + (size_t)BB * NN * HH;           // [B, H, N]
  _Float16* o16  = vT16 + (size_t)BB * NN * HH;           // [B*N, H]

  cvt_f32_f16<<<1024, 256, 0, stream>>>(h,  h16,  BB * NN * MM);
  cvt_f32_f16<<<64,   256, 0, stream>>>(Wq, Wq16, HH * MM);
  cvt_f32_f16<<<64,   256, 0, stream>>>(Wk, Wk16, HH * MM);
  cvt_f32_f16<<<64,   256, 0, stream>>>(Wv, Wv16, HH * MM);
  cvt_f32_f16<<<64,   256, 0, stream>>>(Wo, Wo16, MM * HH);

  const int R = BB * NN;                                   // 16384 rows
  gemm128<0><<<R / 16, 256, 0, stream>>>(h16, Wq16, bq, q16,  nullptr);
  gemm128<0><<<R / 16, 256, 0, stream>>>(h16, Wk16, bk, k16,  nullptr);
  gemm128<1><<<R / 16, 256, 0, stream>>>(h16, Wv16, bv, vT16, nullptr);

  attn_flash<<<dim3(NN / 64, BB), 128, 0, stream>>>(q16, k16, vT16, d, o16);

  gemm128<2><<<R / 16, 256, 0, stream>>>(o16, Wo16, bo, nullptr, out);
}
